// CausalMultiHeadAttention_33148557590898
// MI455X (gfx1250) — matrix operand order repaired and checked by reading, not compiled, not run on hardware
//
#include <hip/hip_runtime.h>

static constexpr int Bb = 4, Tt = 2048, Dd = 2048, Hh = 16, HDd = 128;

typedef __bf16 bf16_t;
typedef __bf16 v16bf __attribute__((ext_vector_type(16)));
typedef __bf16 v8bf  __attribute__((ext_vector_type(8)));
typedef float  v8f   __attribute__((ext_vector_type(8)));
typedef int    v4i   __attribute__((ext_vector_type(4)));

typedef __attribute__((address_space(1))) v4i gv4i;   // global-AS 16B chunk
typedef __attribute__((address_space(3))) v4i sv4i;   // LDS-AS 16B chunk

union BF8 { uint4 u; v8bf v; unsigned short s[8]; };

__device__ __forceinline__ unsigned short f32_to_bf16_bits(float f) {
  unsigned int u = __float_as_uint(f);
  u += 0x7fffu + ((u >> 16) & 1u);   // round-to-nearest-even
  return (unsigned short)(u >> 16);
}

#define WMMA_BF16(Am, Bm, Cm) \
  __builtin_amdgcn_wmma_f32_16x16x32_bf16(false, (Am), false, (Bm), (short)0, (Cm), false, false)

// ---- CDNA5 async copy: 16B per lane, global -> LDS, tracked by ASYNCcnt ----
__device__ __forceinline__ void async_cp16(const bf16_t* gp, bf16_t* sp) {
#if __has_builtin(__builtin_amdgcn_global_load_async_to_lds_b128)
  __builtin_amdgcn_global_load_async_to_lds_b128((gv4i*)gp, (sv4i*)sp, 0, 0);
#else
  unsigned so = (unsigned)(unsigned long long)sp;       // low 32 bits of generic LDS ptr = LDS offset
  unsigned long long ga = (unsigned long long)gp;
  asm volatile("global_load_async_to_lds_b128 %0, %1, off" :: "v"(so), "v"(ga) : "memory");
#endif
}

__device__ __forceinline__ void wait_async0() {
#if __has_builtin(__builtin_amdgcn_s_wait_asynccnt)
  __builtin_amdgcn_s_wait_asynccnt(0);
#else
  asm volatile("s_wait_asynccnt 0" ::: "memory");
#endif
}

// A fragment 16x32 (MxK): lane m holds row m; K = {0..7,16..23} (lanes 0-15)
// or {8..15,24..31} (lanes 16-31). Two 16B loads.
__device__ __forceinline__ v16bf load_frag_A(const bf16_t* base, int ld, int row0, int k0, int lane) {
  const int m = lane & 15, hf = lane >> 4;
  const bf16_t* p = base + (size_t)(row0 + m) * ld + k0 + 8 * hf;
  BF8 lo, hi;
  lo.u = *reinterpret_cast<const uint4*>(p);
  hi.u = *reinterpret_cast<const uint4*>(p + 16);
  v16bf a;
#pragma unroll
  for (int i = 0; i < 8; ++i) { a[i] = lo.v[i]; a[i + 8] = hi.v[i]; }
  return a;
}

// B fragment 32x16 (KxN) where B[k][n] = SRC[n0+n][k0+k] (row-major SRC, rows = B cols).
// lane n holds col n, K={0..7,16..23} (lanes 0-15) or {8..15,24..31} (lanes 16-31). Two 16B loads.
__device__ __forceinline__ v16bf load_frag_B(const bf16_t* base, int ld, int n0, int k0, int lane) {
  const int n = lane & 15, hf = lane >> 4;
  const bf16_t* p = base + (size_t)(n0 + n) * ld + k0 + 8 * hf;
  BF8 lo, hi;
  lo.u = *reinterpret_cast<const uint4*>(p);
  hi.u = *reinterpret_cast<const uint4*>(p + 16);
  v16bf b;
#pragma unroll
  for (int i = 0; i < 8; ++i) { b[i] = lo.v[i]; b[i + 8] = hi.v[i]; }
  return b;
}

__global__ void cvt_f32_bf16(const float* __restrict__ src, unsigned short* __restrict__ dst, int n4) {
  int i = blockIdx.x * 256 + threadIdx.x;
  if (i < n4) {
    float4 f = reinterpret_cast<const float4*>(src)[i];
    ushort4 o;
    o.x = f32_to_bf16_bits(f.x); o.y = f32_to_bf16_bits(f.y);
    o.z = f32_to_bf16_bits(f.z); o.w = f32_to_bf16_bits(f.w);
    reinterpret_cast<ushort4*>(dst)[i] = o;
  }
}

// C = A * W^T with async-DMA double-buffered LDS slabs.
// A: [M x Kdim] bf16 row-major.  W: [N x Kdim] bf16 row-major (rows = out channels).
// mode 0: bf16 out per-head [B,H,T,HD];  mode 1: bf16 out transposed per-head [B,H,HD,T];
// mode 2: f32 out [B,T,D] + bias.
__global__ __launch_bounds__(256) void gemm_wmma(
    const bf16_t* __restrict__ A, const bf16_t* __restrict__ W,
    unsigned short* __restrict__ outb, float* __restrict__ outf,
    const float* __restrict__ bias, int Kdim, int mode)
{
  const int tid  = threadIdx.x;
  const int lane = tid & 31;
  const int wave = tid >> 5;
  const int arow = blockIdx.y * 128;             // block rows in A
  const int wcol = blockIdx.x * 128;             // block rows in W (= output cols)
  const int wm = (wave & 3) * 32;                // wave row offset inside slab
  const int wn = (wave >> 2) * 64;               // wave col offset inside slab

  // double-buffered slabs: 128 rows x 32 k, row-major ld=32 (8KB each)
  __shared__ __align__(16) bf16_t Asl[2][128 * 32];
  __shared__ __align__(16) bf16_t Bsl[2][128 * 32];

  v8f acc[2][4] = {};

  // stage one 32-wide k slab pair: 256 threads x (2 A + 2 B) x 16B async copies
  auto stage = [&](int buf, int k0) {
#pragma unroll
    for (int j = 0; j < 2; ++j) {
      const int c = tid + 256 * j;               // chunk id 0..511; 4 chunks per row
      const int r = c >> 2, o = (c & 3) * 8;     // row, element offset in row
      async_cp16(A + (size_t)(arow + r) * Kdim + k0 + o, &Asl[buf][c * 8]);
      async_cp16(W + (size_t)(wcol + r) * Kdim + k0 + o, &Bsl[buf][c * 8]);
    }
  };

  stage(0, 0);
  for (int k0 = 0; k0 < Kdim; k0 += 32) {
    const int cur = (k0 >> 5) & 1;
    wait_async0();          // this wave's copies into buf[cur] done
    __syncthreads();        // everyone's copies done; everyone's prior reads of buf[cur^1] retired
    if (k0 + 32 < Kdim) stage(cur ^ 1, k0 + 32);

    v16bf a0 = load_frag_A(&Asl[cur][0], 32, wm,      0, lane);
    v16bf a1 = load_frag_A(&Asl[cur][0], 32, wm + 16, 0, lane);
    v16bf b0 = load_frag_B(&Bsl[cur][0], 32, wn +  0, 0, lane);
    v16bf b1 = load_frag_B(&Bsl[cur][0], 32, wn + 16, 0, lane);
    v16bf b2 = load_frag_B(&Bsl[cur][0], 32, wn + 32, 0, lane);
    v16bf b3 = load_frag_B(&Bsl[cur][0], 32, wn + 48, 0, lane);
    acc[0][0] = WMMA_BF16(a0, b0, acc[0][0]);
    acc[0][1] = WMMA_BF16(a0, b1, acc[0][1]);
    acc[0][2] = WMMA_BF16(a0, b2, acc[0][2]);
    acc[0][3] = WMMA_BF16(a0, b3, acc[0][3]);
    acc[1][0] = WMMA_BF16(a1, b0, acc[1][0]);
    acc[1][1] = WMMA_BF16(a1, b1, acc[1][1]);
    acc[1][2] = WMMA_BF16(a1, b2, acc[1][2]);
    acc[1][3] = WMMA_BF16(a1, b3, acc[1][3]);
  }

  const int row0 = arow + wm, col0 = wcol + wn;
  const int n = lane & 15, hf = lane >> 4;
  if (mode == 2) {
#pragma unroll
    for (int j = 0; j < 4; ++j) {
      const int gcol = col0 + 16 * j + n;
      const float bv = bias[gcol];
#pragma unroll
      for (int i = 0; i < 2; ++i)
#pragma unroll
        for (int r = 0; r < 8; ++r) {
          const int grow = row0 + 16 * i + 8 * hf + r;
          outf[(size_t)grow * Dd + gcol] = acc[i][j][r] + bv;
        }
    }
  } else if (mode == 0) {
#pragma unroll
    for (int j = 0; j < 4; ++j) {
      const int gcol = col0 + 16 * j + n;
      const int h = gcol >> 7, hd = gcol & (HDd - 1);
#pragma unroll
      for (int i = 0; i < 2; ++i)
#pragma unroll
        for (int r = 0; r < 8; ++r) {
          const int grow = row0 + 16 * i + 8 * hf + r;
          const int bidx = grow >> 11, t = grow & (Tt - 1);
          outb[(((size_t)(bidx * Hh + h)) * Tt + t) * HDd + hd] = f32_to_bf16_bits(acc[i][j][r]);
        }
    }
  } else {  // mode 1: transposed per-head store; lane writes 8 contiguous t's -> one b128 store
#pragma unroll
    for (int j = 0; j < 4; ++j) {
      const int gcol = col0 + 16 * j + n;
      const int h = gcol >> 7, hd = gcol & (HDd - 1);
#pragma unroll
      for (int i = 0; i < 2; ++i) {
        const int t0 = row0 + 16 * i + 8 * hf;
        const int bidx = t0 >> 11, t = t0 & (Tt - 1);
        BF8 pk;
#pragma unroll
        for (int r = 0; r < 8; ++r) pk.s[r] = f32_to_bf16_bits(acc[i][j][r]);
        *reinterpret_cast<uint4*>(&outb[(((size_t)(bidx * Hh + h)) * HDd + hd) * Tt + t]) = pk.u;
      }
    }
  }
}

// Flash attention: one wave per (b,h,16-query tile). Single-wave workgroup so barriers are NOPs
// and causal trip-count divergence across workgroups is safe.
__global__ __launch_bounds__(32) void attn_wmma(
    const bf16_t* __restrict__ Q, const bf16_t* __restrict__ K,
    const bf16_t* __restrict__ Vt, unsigned short* __restrict__ ctx)
{
  const int lane = threadIdx.x;
  const int qt = blockIdx.x & (Tt / 16 - 1);
  const int bh = blockIdx.x >> 7;
  const int qbase = qt * 16;
  const bf16_t* Qp = Q  + (size_t)bh * Tt * HDd;
  const bf16_t* Kp = K  + (size_t)bh * Tt * HDd;
  const bf16_t* Vp = Vt + (size_t)bh * HDd * Tt;

  v16bf qf[4];
#pragma unroll
  for (int s = 0; s < 4; ++s) qf[s] = load_frag_A(Qp, HDd, qbase, 32 * s, lane);

  v8f acc[8] = {};
  float mrow[8], lrow[8];
#pragma unroll
  for (int r = 0; r < 8; ++r) { mrow[r] = -3.0e38f; lrow[r] = 0.f; }

  const int n = lane & 15, hf = lane >> 4;
  const int moff = 8 * hf;
  const float scale = 0.08838834764831845f;  // 1/sqrt(128)

  __shared__ __align__(16) unsigned short Pl[16 * 32];

  const int nkt = (qbase + 47) >> 5;  // causal: tiles covering keys 0..qbase+15
  for (int kt = 0; kt < nkt; ++kt) {
    const int kb = kt * 32;
    v8f s0 = {}, s1 = {};
#pragma unroll
    for (int s = 0; s < 4; ++s) {
      v16bf k0f = load_frag_B(Kp, HDd, kb,      32 * s, lane);
      v16bf k1f = load_frag_B(Kp, HDd, kb + 16, 32 * s, lane);
      s0 = WMMA_BF16(qf[s], k0f, s0);
      s1 = WMMA_BF16(qf[s], k1f, s1);
    }
    float corr[8];
#pragma unroll
    for (int r = 0; r < 8; ++r) {
      const int q = qbase + moff + r;
      float v0 = (kb + n      <= q) ? s0[r] * scale : -3.0e38f;
      float v1 = (kb + 16 + n <= q) ? s1[r] * scale : -3.0e38f;
      float mx = fmaxf(v0, v1);
      mx = fmaxf(mx, __shfl_xor(mx, 1, 32));
      mx = fmaxf(mx, __shfl_xor(mx, 2, 32));
      mx = fmaxf(mx, __shfl_xor(mx, 4, 32));
      mx = fmaxf(mx, __shfl_xor(mx, 8, 32));
      const float mn = fmaxf(mrow[r], mx);
      corr[r] = __expf(mrow[r] - mn);
      mrow[r] = mn;
      const float p0 = __expf(v0 - mn);
      const float p1 = __expf(v1 - mn);
      s0[r] = p0; s1[r] = p1;
      float rs = p0 + p1;
      rs += __shfl_xor(rs, 1, 32);
      rs += __shfl_xor(rs, 2, 32);
      rs += __shfl_xor(rs, 4, 32);
      rs += __shfl_xor(rs, 8, 32);
      lrow[r] = lrow[r] * corr[r] + rs;
    }
#pragma unroll
    for (int f = 0; f < 8; ++f)
#pragma unroll
      for (int r = 0; r < 8; ++r) acc[f][r] *= corr[r];

    __syncthreads();  // single-wave WG: ensures prior LDS reads retired before overwrite
#pragma unroll
    for (int r = 0; r < 8; ++r) {
      Pl[(moff + r) * 32 + n]      = f32_to_bf16_bits(s0[r]);
      Pl[(moff + r) * 32 + 16 + n] = f32_to_bf16_bits(s1[r]);
    }
    __syncthreads();
    v16bf pf = load_frag_A(reinterpret_cast<const bf16_t*>(Pl), 32, 0, 0, lane);  // C->A relayout
#pragma unroll
    for (int f = 0; f < 8; ++f) {
      v16bf vf = load_frag_B(Vp, Tt, f * 16, kb, lane);  // Vt rows are B columns
      acc[f] = WMMA_BF16(pf, vf, acc[f]);
    }
  }

  const int h = bh & (Hh - 1), bidx = bh >> 4;
#pragma unroll
  for (int f = 0; f < 8; ++f) {
    const int gcol = h * HDd + f * 16 + n;
#pragma unroll
    for (int r = 0; r < 8; ++r) {
      const int q = qbase + moff + r;
      ctx[((size_t)(bidx * Tt + q)) * Dd + gcol] = f32_to_bf16_bits(acc[f][r] / lrow[r]);
    }
  }
}

extern "C" void kernel_launch(void* const* d_in, const int* in_sizes, int n_in,
                              void* d_out, int out_size, void* d_ws, size_t ws_size,
                              hipStream_t stream) {
  const float* x  = (const float*)d_in[0];
  const float* Wq = (const float*)d_in[1];
  const float* Wk = (const float*)d_in[2];
  const float* Wv = (const float*)d_in[3];
  const float* Wo = (const float*)d_in[4];
  const float* bo = (const float*)d_in[5];
  float* out = (float*)d_out;

  const size_t nx = (size_t)Bb * Tt * Dd;  // 16M elements
  const size_t nw = (size_t)Dd * Dd;       // 4M elements

  char* ws = (char*)d_ws;
  unsigned short* xb   = (unsigned short*)ws; ws += nx * 2;
  unsigned short* wqb  = (unsigned short*)ws; ws += nw * 2;
  unsigned short* wkb  = (unsigned short*)ws; ws += nw * 2;
  unsigned short* wvb  = (unsigned short*)ws; ws += nw * 2;
  unsigned short* wob  = (unsigned short*)ws; ws += nw * 2;
  unsigned short* Qh   = (unsigned short*)ws; ws += nx * 2;
  unsigned short* Kh   = (unsigned short*)ws; ws += nx * 2;
  unsigned short* Vth  = (unsigned short*)ws; ws += nx * 2;
  unsigned short* ctxb = (unsigned short*)ws; ws += nx * 2;

  auto cvt = [&](const float* s, unsigned short* d, size_t cnt) {
    int n4 = (int)(cnt / 4);
    cvt_f32_bf16<<<dim3((n4 + 255) / 256), dim3(256), 0, stream>>>(s, d, n4);
  };
  cvt(x, xb, nx);
  cvt(Wq, wqb, nw); cvt(Wk, wkb, nw); cvt(Wv, wvb, nw); cvt(Wo, wob, nw);

  dim3 ggrid(Dd / 128, (Bb * Tt) / 128);  // 16 x 64
  gemm_wmma<<<ggrid, 256, 0, stream>>>((const bf16_t*)xb, (const bf16_t*)wqb, Qh,  nullptr, nullptr, Dd, 0);
  gemm_wmma<<<ggrid, 256, 0, stream>>>((const bf16_t*)xb, (const bf16_t*)wkb, Kh,  nullptr, nullptr, Dd, 0);
  gemm_wmma<<<ggrid, 256, 0, stream>>>((const bf16_t*)xb, (const bf16_t*)wvb, Vth, nullptr, nullptr, Dd, 1);

  attn_wmma<<<dim3(Bb * Hh * (Tt / 16)), dim3(32), 0, stream>>>(
      (const bf16_t*)Qh, (const bf16_t*)Kh, (const bf16_t*)Vth, ctxb);

  gemm_wmma<<<ggrid, 256, 0, stream>>>((const bf16_t*)ctxb, (const bf16_t*)wob, nullptr, out, bo, Dd, 2);
}
